// MambaBlock_76098230550867
// MI455X (gfx1250) — matrix operand order repaired and checked by reading, not compiled, not run on hardware
//
#include <hip/hip_runtime.h>
#include <hip/hip_bf16.h>
#include <math.h>

// ---------------------------------------------------------------------------
// MambaBlock for MI455X (gfx1250). All pointwise GEMMs via
// v_wmma_f32_16x16x32_bf16 (bf16 in, f32 accumulate); selective scan as a
// 3-pass chunked scan for parallelism. fp32 intermediates in a d_ws arena.
// Round 2: branch-free WMMA inner loops (templated tile counts, select-based
// weight-row guard) and 2 M-tiles per wave in the sequence GEMMs.
// ---------------------------------------------------------------------------

typedef __attribute__((ext_vector_type(16))) __bf16 v16bf;
typedef __attribute__((ext_vector_type(8)))  float  v8f;

#define BB   8
#define CCh  64
#define HHt  128
#define WWd  128
#define HWc  (HHt*WWd)      // 16384
#define LLs  HWc
#define DINc 128
#define DSn  16
#define DTR  4
#define HIDc 128
#define OWq  66             // half-band size for db3 on 128
#define CDIV(a,b) (((a)+(b)-1)/(b))

__constant__ float cDEC_LO[6] = { 0.035226291882100656f, -0.08544127388224149f,
                                 -0.13501102001039084f,  0.4598775021193313f,
                                  0.8068915093133388f,   0.3326705529509569f };
__constant__ float cDEC_HI[6] = {-0.3326705529509569f,   0.8068915093133388f,
                                 -0.4598775021193313f,  -0.13501102001039084f,
                                  0.08544127388224149f,  0.035226291882100656f };

__device__ __forceinline__ float siluf(float v){ return v / (1.f + expf(-v)); }
__device__ __forceinline__ float geluf(float v){ return 0.5f*v*(1.f + erff(v*0.70710678118654752f)); }

// ----------------------------- WMMA helpers --------------------------------
__device__ __forceinline__ v8f wmma_bf16(v16bf a, v16bf b, v8f c){
  return __builtin_amdgcn_wmma_f32_16x16x32_bf16(false, a, false, b, (short)0, c, false, false);
}

// A tile 16x32 (M x K) from row-major fp32 [*, ld]; lane = m + 16*half,
// per-lane 16 contiguous k (k = half*16 + e).
__device__ __forceinline__ v16bf load_a16x32(const float* __restrict__ src, int ld, int m0, int k0){
  int lane = threadIdx.x & 31;
  int m = lane & 15, half = lane >> 4;
  const float* r0 = src + (size_t)(m0 + m)*ld + k0 + half*16;
  v16bf a;
#pragma unroll
  for (int i = 0; i < 8; i++){ a[i] = (__bf16)r0[i]; a[8+i] = (__bf16)r0[8+i]; }
  return a;
}

// B tile 32x16 (K x N) where B[k,n] = W[n0+n, k0+k], W row-major.
// lane: n = lane&15, per-lane 16 contiguous k (k = half*16 + e). No guard.
__device__ __forceinline__ v16bf load_b_w(const float* __restrict__ Wt, int ld, int n0, int k0){
  int lane = threadIdx.x & 31;
  int n = n0 + (lane & 15), half = lane >> 4;
  const float* r = Wt + (size_t)n*ld + k0 + half*16;
  v16bf b;
#pragma unroll
  for (int e = 0; e < 16; e++) b[e] = (__bf16)r[e];
  return b;
}

// Branch-free guarded variant (rows >= Nw contribute zero): clamp + select.
__device__ __forceinline__ v16bf load_b_w_guard(const float* __restrict__ Wt, int ld,
                                                int n0, int k0, int Nw){
  int lane = threadIdx.x & 31;
  int n = n0 + (lane & 15), half = lane >> 4;
  int nc = (n < Nw) ? n : (Nw - 1);
  float sel = (n < Nw) ? 1.f : 0.f;
  const float* r = Wt + (size_t)nc*ld + k0 + half*16;
  v16bf b;
#pragma unroll
  for (int e = 0; e < 16; e++) b[e] = (__bf16)(r[e] * sel);
  return b;
}

// B tile 32x16 where B[k,n] = In[(k0+k)*ldrow + hw0+n] (NCHW channel-mix).
__device__ __forceinline__ v16bf load_b_act(const float* __restrict__ src, int ldrow, int hw0, int k0){
  int lane = threadIdx.x & 31;
  int n = hw0 + (lane & 15), half = lane >> 4;
  const float* r = src + (size_t)(k0 + half*16)*ldrow + n;
  v16bf b;
#pragma unroll
  for (int e = 0; e < 16; e++) b[e] = (__bf16)r[(size_t)e * ldrow];
  return b;
}

// --------------------- G1: sequence GEMM Out = In * W^T ---------------------
// In [M,K] row-major, W [Nw,K] row-major, Out [M,Npad=NT*16*ngroups].
// One wave: 32 (two M tiles) x NT*16 output, B tiles reused across both A's.
template<int NT, bool GUARD>
__global__ __launch_bounds__(256)
void k_gemm_seq(const float* __restrict__ In, const float* __restrict__ Wt,
                float* __restrict__ Out, int M, int K, int Npad, int Nw,
                int ngroups, int totalWaves){
  int wid = blockIdx.x * (blockDim.x >> 5) + (threadIdx.x >> 5);
  if (wid >= totalWaves) return;
  int mt = wid / ngroups, ng = wid % ngroups;
  int m0 = mt * 32, n0g = ng * (NT * 16);
  v8f acc0[NT], acc1[NT];
#pragma unroll
  for (int j = 0; j < NT; j++)
#pragma unroll
    for (int r = 0; r < 8; r++){ acc0[j][r] = 0.f; acc1[j][r] = 0.f; }
  for (int k0 = 0; k0 < K; k0 += 32){
    v16bf a0 = load_a16x32(In, K, m0,      k0);
    v16bf a1 = load_a16x32(In, K, m0 + 16, k0);
#pragma unroll
    for (int j = 0; j < NT; j++){
      v16bf b = GUARD ? load_b_w_guard(Wt, K, n0g + j*16, k0, Nw)
                      : load_b_w      (Wt, K, n0g + j*16, k0);
      acc0[j] = wmma_bf16(a0, b, acc0[j]);
      acc1[j] = wmma_bf16(a1, b, acc1[j]);
    }
  }
  int lane = threadIdx.x & 31;
  int n = lane & 15, half = lane >> 4;
#pragma unroll
  for (int j = 0; j < NT; j++){
    int nc = n0g + j*16 + n;
#pragma unroll
    for (int r = 0; r < 8; r++){
      Out[(size_t)(m0 +      r + half*8)*Npad + nc] = acc0[j][r];
      Out[(size_t)(m0 + 16 + r + half*8)*Npad + nc] = acc1[j][r];
    }
  }
}

// ------------- G2: NCHW channel mix Out[b,o,hw] = sum_c W[o,c] In[b,c,hw] ---
__global__ __launch_bounds__(256)
void k_gemm_chan(const float* __restrict__ In, const float* __restrict__ Wt,
                 float* __restrict__ Out, const float* __restrict__ Add,
                 int K, int O, int totalWaves){
  int wid = blockIdx.x * (blockDim.x >> 5) + (threadIdx.x >> 5);
  if (wid >= totalWaves) return;
  const int ngr = HWc / 64;                 // 256 hw-groups
  int hg = wid % ngr; int t = wid / ngr;
  int ot = t % (O / 16); int b = t / (O / 16);
  const float* inb = In + (size_t)b * K * HWc;
  int o0 = ot * 16, hw0g = hg * 64;
  v8f acc[4];
#pragma unroll
  for (int j = 0; j < 4; j++)
#pragma unroll
    for (int r = 0; r < 8; r++) acc[j][r] = 0.f;
  for (int k0 = 0; k0 < K; k0 += 32){
    v16bf a = load_a16x32(Wt, K, o0, k0);   // weights as A matrix
#pragma unroll
    for (int j = 0; j < 4; j++){
      v16bf bm = load_b_act(inb, HWc, hw0g + j*16, k0);
      acc[j] = wmma_bf16(a, bm, acc[j]);
    }
  }
  int lane = threadIdx.x & 31;
  int n = lane & 15, half = lane >> 4;
#pragma unroll
  for (int j = 0; j < 4; j++){
#pragma unroll
    for (int r = 0; r < 8; r++){
      size_t idx = ((size_t)b*O + o0 + r + half*8)*HWc + hw0g + j*16 + n;
      float v = acc[j][r];
      if (Add) v += Add[idx];
      Out[idx] = v;
    }
  }
}

// ------------------------- LayerNorm over channels --------------------------
// mode 0: out NCHW. mode 1: out [b, hw, C] (row-major, C contiguous).
__global__ void k_ln_chan(const float* __restrict__ in, const float* __restrict__ g,
                          const float* __restrict__ bt, float* __restrict__ out, int mode){
  int idx = blockIdx.x * blockDim.x + threadIdx.x;
  if (idx >= BB * HWc) return;
  int b = idx / HWc, hw = idx % HWc;
  const float* p = in + (size_t)b * CCh * HWc + hw;
  float v[CCh]; float s = 0.f, ss = 0.f;
#pragma unroll
  for (int c = 0; c < CCh; c++){ float u = p[(size_t)c * HWc]; v[c] = u; s += u; ss += u*u; }
  float m = s * (1.f/CCh);
  float var = ss * (1.f/CCh) - m*m;
  float r = rsqrtf(var + 1e-5f);
#pragma unroll
  for (int c = 0; c < CCh; c++){
    float o = (v[c] - m) * r * g[c] + bt[c];
    if (mode == 0) out[((size_t)b*CCh + c)*HWc + hw] = o;
    else           out[((size_t)b*HWc + hw)*CCh + c] = o;
  }
}

// ----------------------------- DWT / IDWT ----------------------------------
__global__ void k_dwt_row(const float* __restrict__ in, float* __restrict__ lo,
                          float* __restrict__ hi){
  int idx = blockIdx.x * blockDim.x + threadIdx.x;
  const int N = BB*CCh*HHt*OWq;
  if (idx >= N) return;
  int ow = idx % OWq; int t1 = idx / OWq;
  int h = t1 % HHt;   int bc = t1 / HHt;
  const float* row = in + ((size_t)bc*HHt + h)*WWd;
  float aL = 0.f, aH = 0.f;
#pragma unroll
  for (int t = 0; t < 6; t++){
    int x = 2*ow + t - 4;
    if (x >= 0 && x < WWd){ float u = row[x]; aL += u*cDEC_LO[5-t]; aH += u*cDEC_HI[5-t]; }
  }
  size_t o = ((size_t)bc*HHt + h)*OWq + ow;
  lo[o] = aL; hi[o] = aH;
}

__global__ void k_dwt_col(const float* __restrict__ in, float* __restrict__ oL,
                          float* __restrict__ oH){
  int idx = blockIdx.x * blockDim.x + threadIdx.x;
  const int N = BB*CCh*OWq*OWq;
  if (idx >= N) return;
  int wp = idx % OWq; int t1 = idx / OWq;
  int oh = t1 % OWq;  int bc = t1 / OWq;
  const float* base = in + (size_t)bc*HHt*OWq;
  float aL = 0.f, aH = 0.f;
#pragma unroll
  for (int t = 0; t < 6; t++){
    int y = 2*oh + t - 4;
    if (y >= 0 && y < HHt){ float u = base[(size_t)y*OWq + wp]; aL += u*cDEC_LO[5-t]; aH += u*cDEC_HI[5-t]; }
  }
  size_t o = ((size_t)bc*OWq + oh)*OWq + wp;
  oL[o] = aL; oH[o] = aH;
}

// lo2[b,c,128,66] = uph(aL, DEC_LO) + uph(aH, DEC_HI)  (stride-2 upsample in H)
__global__ void k_idwt_col(const float* __restrict__ aL, const float* __restrict__ aH,
                           float* __restrict__ out){
  int idx = blockIdx.x * blockDim.x + threadIdx.x;
  const int N = BB*CCh*HHt*OWq;
  if (idx >= N) return;
  int wp = idx % OWq; int t1 = idx / OWq;
  int oh = t1 % HHt;  int bc = t1 / HHt;
  const float* bL = aL + (size_t)bc*OWq*OWq;
  const float* bH = aH + (size_t)bc*OWq*OWq;
  float acc = 0.f;
#pragma unroll
  for (int t = 0; t < 6; t++){
    int j = oh + t - 1;                 // dilated+padded index
    if (j >= 0 && (j & 1) == 0){
      int i = j >> 1;
      if (i < OWq){
        size_t s = (size_t)i*OWq + wp;
        acc += bL[s]*cDEC_LO[t] + bH[s]*cDEC_HI[t];
      }
    }
  }
  out[((size_t)bc*HHt + oh)*OWq + wp] = acc;
}

__global__ void k_idwt_row(const float* __restrict__ lo2, const float* __restrict__ hi2,
                           float* __restrict__ q){
  int idx = blockIdx.x * blockDim.x + threadIdx.x;
  const int N = BB*CCh*HHt*WWd;
  if (idx >= N) return;
  int x = idx % WWd; int t1 = idx / WWd;
  int y = t1 % HHt;  int bc = t1 / HHt;
  const float* rL = lo2 + ((size_t)bc*HHt + y)*OWq;
  const float* rH = hi2 + ((size_t)bc*HHt + y)*OWq;
  float acc = 0.f;
#pragma unroll
  for (int t = 0; t < 6; t++){
    int j = x + t - 1;
    if (j >= 0 && (j & 1) == 0){
      int i = j >> 1;
      if (i < OWq) acc += rL[i]*cDEC_LO[t] + rH[i]*cDEC_HI[t];
    }
  }
  q[((size_t)bc*HHt + y)*WWd + x] = acc;
}

// ------------------------ 3x3 depthwise conv (pad 1) ------------------------
__global__ void k_conv3x3_dw(const float* __restrict__ in, const float* __restrict__ w,
                             float* __restrict__ out, int CH, int Hh, int Wv, int addIdent){
  int idx = blockIdx.x * blockDim.x + threadIdx.x;
  int N = BB*CH*Hh*Wv;
  if (idx >= N) return;
  int x = idx % Wv; int t1 = idx / Wv;
  int y = t1 % Hh;  int t2 = t1 / Hh;
  int c = t2 % CH;
  const float* base = in + (size_t)t2*Hh*Wv;
  const float* wk = w + c*9;
  float acc = 0.f;
#pragma unroll
  for (int i = 0; i < 3; i++){
    int yy = y - 1 + i; if (yy < 0 || yy >= Hh) continue;
#pragma unroll
    for (int j = 0; j < 3; j++){
      int xx = x - 1 + j; if (xx < 0 || xx >= Wv) continue;
      acc += base[(size_t)yy*Wv + xx] * wk[i*3 + j];
    }
  }
  float idv = addIdent ? base[(size_t)y*Wv + x] : 0.f;
  out[idx] = acc + idv;
}

// ------------------- split xz -> xm(snake order) and z ----------------------
__global__ void k_split_snake(const float* __restrict__ xz, float* __restrict__ xm,
                              float* __restrict__ z){
  size_t idx = (size_t)blockIdx.x * blockDim.x + threadIdx.x;
  const size_t N = (size_t)BB*LLs*DINc;
  if (idx >= N) return;
  int d = idx % DINc; size_t t1 = idx / DINc;
  int l = t1 % LLs;   int b = t1 / LLs;
  int h = l / WWd, wv = l % WWd;
  int lsrc = (h & 1) ? (h*WWd + (WWd-1 - wv)) : l;
  xm[idx] = xz[((size_t)b*LLs + lsrc)*256 + d];
  z [idx] = xz[((size_t)b*LLs + l   )*256 + 128 + d];
}

// ---------------- causal depthwise conv1d (k=4) + bias + silu ---------------
__global__ void k_conv1d_silu(const float* __restrict__ xm, const float* __restrict__ w,
                              const float* __restrict__ bias, float* __restrict__ out){
  size_t idx = (size_t)blockIdx.x * blockDim.x + threadIdx.x;
  const size_t N = (size_t)BB*LLs*DINc;
  if (idx >= N) return;
  int d = idx % DINc; size_t t1 = idx / DINc;
  int l = t1 % LLs;   int b = t1 / LLs;
  float acc = bias[d];
  const float* wd = w + d*4;
#pragma unroll
  for (int t = 0; t < 4; t++){
    int ls = l - 3 + t;
    if (ls >= 0) acc += xm[((size_t)b*LLs + ls)*DINc + d] * wd[t];
  }
  out[idx] = siluf(acc);
}

// ------------------------ chunked selective scan ---------------------------
#define NCH 128
#define CLN 128
// pass 1: per-(b,chunk) scan with zero init; store per-l ys & inclusive cumdt,
// per-chunk end state & chunk dt-sum.
__global__ __launch_bounds__(128)
void k_scan1(const float* __restrict__ dbc, const float* __restrict__ xt,
             const float* __restrict__ dtw, const float* __restrict__ dtb,
             const float* __restrict__ Alog, float* __restrict__ ys,
             float* __restrict__ cum, float* __restrict__ hbuf,
             float* __restrict__ ccum){
  int b = blockIdx.x / NCH, ch = blockIdx.x % NCH;
  int d = threadIdx.x;
  float A[DSn];
#pragma unroll
  for (int s = 0; s < DSn; s++) A[s] = -expf(Alog[d*DSn + s]);
  float w0 = dtw[d*4+0], w1 = dtw[d*4+1], w2 = dtw[d*4+2], w3 = dtw[d*4+3];
  float bconst = dtb[d];
  float h[DSn];
#pragma unroll
  for (int s = 0; s < DSn; s++) h[s] = 0.f;
  float cs = 0.f;
  for (int li = 0; li < CLN; li++){
    int l = ch*CLN + li;
    const float* row = dbc + ((size_t)b*LLs + l)*48;
    float u = row[0]*w0 + row[1]*w1 + row[2]*w2 + row[3]*w3 + bconst;
    float dt = (u > 20.f) ? u : log1pf(expf(u));
    float x = xt[((size_t)b*LLs + l)*DINc + d];
    float dtx = dt * x;
    float y = 0.f;
#pragma unroll
    for (int s = 0; s < DSn; s++){
      float e = expf(dt * A[s]);
      h[s] = h[s]*e + dtx*row[4 + s];
      y += h[s]*row[20 + s];
    }
    cs += dt;
    size_t o = ((size_t)b*LLs + l)*DINc + d;
    ys[o] = y; cum[o] = cs;
  }
  size_t hb = (((size_t)b*NCH + ch)*DINc + d)*DSn;
#pragma unroll
  for (int s = 0; s < DSn; s++) hbuf[hb + s] = h[s];
  ccum[((size_t)b*NCH + ch)*DINc + d] = cs;
}

// pass 2: serial over 128 chunks per (b,d); replaces hbuf with EXCLUSIVE init.
__global__ __launch_bounds__(128)
void k_scan2(float* __restrict__ hbuf, const float* __restrict__ ccum,
             const float* __restrict__ Alog){
  int b = blockIdx.x, d = threadIdx.x;
  float A[DSn];
#pragma unroll
  for (int s = 0; s < DSn; s++) A[s] = -expf(Alog[d*DSn + s]);
  float carry[DSn];
#pragma unroll
  for (int s = 0; s < DSn; s++) carry[s] = 0.f;
  for (int ch = 0; ch < NCH; ch++){
    size_t hb = (((size_t)b*NCH + ch)*DINc + d)*DSn;
    float cs = ccum[((size_t)b*NCH + ch)*DINc + d];
#pragma unroll
    for (int s = 0; s < DSn; s++){
      float hend = hbuf[hb + s];
      hbuf[hb + s] = carry[s];
      carry[s] = carry[s]*expf(A[s]*cs) + hend;
    }
  }
}

// pass 3: ys += C_l . (h_init(chunk) * exp(A * cumdt_l))
__global__ void k_scan3(float* __restrict__ ys, const float* __restrict__ dbc,
                        const float* __restrict__ cum, const float* __restrict__ hbuf,
                        const float* __restrict__ Alog){
  size_t idx = (size_t)blockIdx.x * blockDim.x + threadIdx.x;
  const size_t N = (size_t)BB*LLs*DINc;
  if (idx >= N) return;
  int d = idx % DINc; size_t t1 = idx / DINc;
  int l = t1 % LLs;   int b = t1 / LLs;
  int ch = l / CLN;
  if (ch == 0) return;                       // h_init is zero there
  const float* row = dbc + ((size_t)b*LLs + l)*48;
  size_t hb = (((size_t)b*NCH + ch)*DINc + d)*DSn;
  float cs = cum[idx];
  float acc = 0.f;
#pragma unroll
  for (int s = 0; s < DSn; s++){
    float A = -expf(Alog[d*DSn + s]);
    acc += row[20 + s] * hbuf[hb + s] * expf(A*cs);
  }
  ys[idx] += acc;
}

// y = snake^-1(ys + xm*D) * silu(z)   (output in raster order)
__global__ void k_ycombine(const float* __restrict__ ys, const float* __restrict__ xm,
                           const float* __restrict__ z, const float* __restrict__ Dw,
                           float* __restrict__ out){
  size_t idx = (size_t)blockIdx.x * blockDim.x + threadIdx.x;
  const size_t N = (size_t)BB*LLs*DINc;
  if (idx >= N) return;
  int d = idx % DINc; size_t t1 = idx / DINc;
  int l = t1 % LLs;   int b = t1 / LLs;
  int h = l / WWd, wv = l % WWd;
  int lp = (h & 1) ? (h*WWd + (WWd-1 - wv)) : l;
  size_t sidx = ((size_t)b*LLs + lp)*DINc + d;
  float y = ys[sidx] + xm[sidx]*Dw[d];
  out[idx] = y * siluf(z[idx]);
}

// [b,hw,64] -> [b,64,hw]
__global__ void k_to_nchw(const float* __restrict__ in, float* __restrict__ out){
  size_t idx = (size_t)blockIdx.x * blockDim.x + threadIdx.x;
  const size_t N = (size_t)BB*CCh*HWc;
  if (idx >= N) return;
  int hw = idx % HWc; size_t t1 = idx / HWc;
  int c = t1 % CCh;   int b = t1 / CCh;
  out[idx] = in[((size_t)b*HWc + hw)*CCh + c];
}

// GDFN gate: g[b,hc,hw] = gelu(dw3x3(t, ch=hc)) * dw3x3(t, ch=hc+128)
__global__ void k_gdfn_gate(const float* __restrict__ t, const float* __restrict__ w,
                            float* __restrict__ g){
  size_t idx = (size_t)blockIdx.x * blockDim.x + threadIdx.x;
  const size_t N = (size_t)BB*HIDc*HWc;
  if (idx >= N) return;
  int x = idx % WWd; size_t t1 = idx / WWd;
  int y = t1 % HHt;  size_t t2 = t1 / HHt;
  int hc = t2 % HIDc; int b = t2 / HIDc;
  const float* p1 = t + ((size_t)b*256 + hc      )*HWc;
  const float* p2 = t + ((size_t)b*256 + hc + 128)*HWc;
  const float* w1 = w + hc*9;
  const float* w2 = w + (hc + 128)*9;
  float a1 = 0.f, a2 = 0.f;
#pragma unroll
  for (int i = 0; i < 3; i++){
    int yy = y - 1 + i; if (yy < 0 || yy >= HHt) continue;
#pragma unroll
    for (int j = 0; j < 3; j++){
      int xx = x - 1 + j; if (xx < 0 || xx >= WWd) continue;
      size_t s = (size_t)yy*WWd + xx;
      a1 += p1[s]*w1[i*3 + j];
      a2 += p2[s]*w2[i*3 + j];
    }
  }
  g[idx] = geluf(a1) * a2;
}

// ---------------------------------------------------------------------------
extern "C" void kernel_launch(void* const* d_in, const int* in_sizes, int n_in,
                              void* d_out, int out_size, void* d_ws, size_t ws_size,
                              hipStream_t stream) {
  const float* x        = (const float*)d_in[0];
  const float* norm1_g  = (const float*)d_in[1];
  const float* norm1_b  = (const float*)d_in[2];
  const float* norm2_g  = (const float*)d_in[3];
  const float* norm2_b  = (const float*)d_in[4];
  const float* ssl_w5   = (const float*)d_in[5];
  const float* ssl_w7   = (const float*)d_in[6];
  const float* ssl_w9   = (const float*)d_in[7];
  const float* aln_g    = (const float*)d_in[8];
  const float* aln_b    = (const float*)d_in[9];
  const float* in_proj  = (const float*)d_in[10];   // [256,64]
  const float* c1d_w    = (const float*)d_in[11];   // [128,1,4]
  const float* c1d_b    = (const float*)d_in[12];
  const float* x_proj   = (const float*)d_in[13];   // [36,128]
  const float* dt_w     = (const float*)d_in[14];   // [128,4]
  const float* dt_b     = (const float*)d_in[15];
  const float* A_log    = (const float*)d_in[16];   // [128,16]
  const float* Dw       = (const float*)d_in[17];
  const float* out_proj = (const float*)d_in[18];   // [64,128]
  const float* lconv_w  = (const float*)d_in[19];   // [64,1,3,3]
  const float* attn_w   = (const float*)d_in[20];   // [64,64]
  const float* gin_w    = (const float*)d_in[21];   // [256,64]
  const float* gdw_w    = (const float*)d_in[22];   // [256,1,3,3]
  const float* gout_w   = (const float*)d_in[23];   // [64,128]
  float* out = (float*)d_out;
  float* ws  = (float*)d_ws;

  // ---- fp32 arena (elements). Total ~169.2M floats (~677 MB). ----
  const size_t S_full = (size_t)BB*CCh*HHt*WWd;     // 8.39M
  const size_t S_half = (size_t)BB*CCh*HHt*OWq;     // 4.33M
  const size_t S_quar = (size_t)BB*CCh*OWq*OWq;     // 2.23M
  const size_t M      = (size_t)BB*LLs;             // 131072 rows
  size_t o_a    = 0;                 // xn -> q -> xn2
  size_t o_lo   = o_a    + S_full;   // lo -> lo2
  size_t o_hi   = o_lo   + S_half;   // hi -> hi2
  size_t o_sub  = o_hi   + S_half;   // ll,lh,hl,hh -> o2 (NCHW)
  size_t o_csub = o_sub  + 4*S_quar; // conv subbands -> o2b
  size_t o_s64  = o_csub + 4*S_quar; // xseq -> o
  size_t o_xz   = o_s64  + M*64;     // xz[M,256] -> gdfn t[b,256,hw]
  size_t o_xm   = o_xz   + M*256;    // xm(snake) -> y_gated(raster)
  size_t o_xc   = o_xm   + M*128;    // xm_s (x_t) -> gdfn gate
  size_t o_z    = o_xc   + M*128;
  size_t o_ys   = o_z    + M*128;
  size_t o_cum  = o_ys   + M*128;
  size_t o_dbc  = o_cum  + M*128;    // [M,48]
  size_t o_hbuf = o_dbc  + M*48;     // [b,128,128,16]
  size_t o_ccum = o_hbuf + (size_t)BB*NCH*DINc*DSn;
  (void)ws_size; (void)in_sizes; (void)n_in; (void)out_size;

  const int T = 256;
  // ---- MambaAttention ----
  k_ln_chan<<<CDIV(BB*HWc, T), T, 0, stream>>>(x, norm1_g, norm1_b, ws + o_a, 0);
  k_dwt_row<<<CDIV(BB*CCh*HHt*OWq, T), T, 0, stream>>>(ws + o_a, ws + o_lo, ws + o_hi);
  k_dwt_col<<<CDIV(BB*CCh*OWq*OWq, T), T, 0, stream>>>(ws + o_lo, ws + o_sub,            ws + o_sub +   S_quar);
  k_dwt_col<<<CDIV(BB*CCh*OWq*OWq, T), T, 0, stream>>>(ws + o_hi, ws + o_sub + 2*S_quar, ws + o_sub + 3*S_quar);
  k_conv3x3_dw<<<CDIV(BB*CCh*OWq*OWq, T), T, 0, stream>>>(ws + o_sub,            ssl_w5, ws + o_csub,            CCh, OWq, OWq, 0);
  k_conv3x3_dw<<<CDIV(BB*CCh*OWq*OWq, T), T, 0, stream>>>(ws + o_sub +   S_quar, ssl_w5, ws + o_csub +   S_quar, CCh, OWq, OWq, 0);
  k_conv3x3_dw<<<CDIV(BB*CCh*OWq*OWq, T), T, 0, stream>>>(ws + o_sub + 2*S_quar, ssl_w7, ws + o_csub + 2*S_quar, CCh, OWq, OWq, 0);
  k_conv3x3_dw<<<CDIV(BB*CCh*OWq*OWq, T), T, 0, stream>>>(ws + o_sub + 3*S_quar, ssl_w9, ws + o_csub + 3*S_quar, CCh, OWq, OWq, 0);
  k_idwt_col<<<CDIV(BB*CCh*HHt*OWq, T), T, 0, stream>>>(ws + o_csub,            ws + o_csub +   S_quar, ws + o_lo);
  k_idwt_col<<<CDIV(BB*CCh*HHt*OWq, T), T, 0, stream>>>(ws + o_csub + 2*S_quar, ws + o_csub + 3*S_quar, ws + o_hi);
  k_idwt_row<<<CDIV(BB*CCh*HHt*WWd, T), T, 0, stream>>>(ws + o_lo, ws + o_hi, ws + o_a);   // q
  k_ln_chan<<<CDIV(BB*HWc, T), T, 0, stream>>>(ws + o_a, aln_g, aln_b, ws + o_s64, 1);     // xseq [b,l,64]

  // in_proj: [M,64] x [256,64]^T -> xz [M,256]   (wave: 32x64, 4 n-groups)
  { int ng = 4, tw = (int)(M/32)*ng;
    k_gemm_seq<4,false><<<tw/8, T, 0, stream>>>(ws + o_s64, in_proj, ws + o_xz, (int)M, 64, 256, 256, ng, tw); }
  k_split_snake<<<CDIV((int)(M*128), T), T, 0, stream>>>(ws + o_xz, ws + o_xm, ws + o_z);
  k_conv1d_silu<<<CDIV((int)(M*128), T), T, 0, stream>>>(ws + o_xm, c1d_w, c1d_b, ws + o_xc);
  // x_proj: [M,128] x [36,128]^T -> dbc [M,48] (rows 36..47 of W read as zero)
  { int ng = 1, tw = (int)(M/32);
    k_gemm_seq<3,true><<<tw/8, T, 0, stream>>>(ws + o_xc, x_proj, ws + o_dbc, (int)M, 128, 48, 36, ng, tw); }
  // chunked selective scan
  k_scan1<<<BB*NCH, 128, 0, stream>>>(ws + o_dbc, ws + o_xc, dt_w, dt_b, A_log,
                                      ws + o_ys, ws + o_cum, ws + o_hbuf, ws + o_ccum);
  k_scan2<<<BB, 128, 0, stream>>>(ws + o_hbuf, ws + o_ccum, A_log);
  k_scan3<<<CDIV((int)(M*128), T), T, 0, stream>>>(ws + o_ys, ws + o_dbc, ws + o_cum, ws + o_hbuf, A_log);
  k_ycombine<<<CDIV((int)(M*128), T), T, 0, stream>>>(ws + o_ys, ws + o_xc, ws + o_z, Dw, ws + o_xm);
  // out_proj: [M,128] x [64,128]^T -> o [M,64]
  { int ng = 1, tw = (int)(M/32);
    k_gemm_seq<4,false><<<tw/8, T, 0, stream>>>(ws + o_xm, out_proj, ws + o_s64, (int)M, 128, 64, 64, ng, tw); }
  k_to_nchw<<<CDIV(BB*CCh*HWc, T), T, 0, stream>>>(ws + o_s64, ws + o_sub);                 // o2
  k_conv3x3_dw<<<CDIV(BB*CCh*HWc, T), T, 0, stream>>>(ws + o_sub, lconv_w, ws + o_csub, CCh, HHt, WWd, 1); // o2b
  // x1 = x + attn_out_w @ o2b  -> d_out
  { int tw = BB*(64/16)*(HWc/64);
    k_gemm_chan<<<tw/8, T, 0, stream>>>(ws + o_csub, attn_w, out, x, 64, 64, tw); }

  // ---- GDFN ----
  k_ln_chan<<<CDIV(BB*HWc, T), T, 0, stream>>>(out, norm2_g, norm2_b, ws + o_a, 0);        // xn2
  { int tw = BB*(256/16)*(HWc/64);
    k_gemm_chan<<<tw/8, T, 0, stream>>>(ws + o_a, gin_w, ws + o_xz, (const float*)nullptr, 64, 256, tw); }
  k_gdfn_gate<<<CDIV(BB*HIDc*HWc, T), T, 0, stream>>>(ws + o_xz, gdw_w, ws + o_xc);
  // d_out += gdfn_out_w @ gate
  { int tw = BB*(64/16)*(HWc/64);
    k_gemm_chan<<<tw/8, T, 0, stream>>>(ws + o_xc, gout_w, out, (const float*)out, 128, 64, tw); }
}
